// DecagonModel_72670846648484
// MI455X (gfx1250) — matrix operand order repaired and checked by reading, not compiled, not run on hardware
//
#include <hip/hip_runtime.h>
#include <stdint.h>

// ---------------------------------------------------------------------------
// Decagon 3-layer hetero-GNN for MI455X (gfx1250, wave32, WMMA).
//   Dense GEMMs (~164 GFLOP)  -> v_wmma_f32_16x16x32_bf16, f32 accumulate.
//     256-thr blocks, 64x128 block tile, 32x32 per wave (4 accumulators,
//     2 A-frags x 2 B-frags -> 4 wmma per K-step), ping-pong LDS A tile,
//     one barrier per K-step.
//   SpMM (2M edges x 3 layers, ~25 GB -> HBM bound at 23.3 TB/s) ->
//     b128 gathers + global_atomic_add_f32 scatter.
// ---------------------------------------------------------------------------

#define NN0 30000
#define NN1 6000
#define FF  1024
#define HH  512
#define M0P 30016   // NN0 padded to 64
#define M1P 6016    // NN1 padded to 64

typedef __attribute__((ext_vector_type(16))) __bf16    bf16x16;
typedef __attribute__((ext_vector_type(8)))  float     f32x8;
typedef __attribute__((ext_vector_type(4)))  uint32_t  u32x4;
typedef __attribute__((ext_vector_type(8)))  uint32_t  u32x8;

__device__ __forceinline__ uint16_t f32_to_bf16_rne(float f) {
    uint32_t u = __float_as_uint(f);
    uint32_t r = u + 0x7fffu + ((u >> 16) & 1u);   // round-to-nearest-even
    return (uint16_t)(r >> 16);
}

// ---------------- elementwise helpers ----------------

// convert nReal elements, zero-fill the padded tail [nReal, nTot)
__global__ void k_f32_to_bf16(const float* __restrict__ in,
                              uint16_t* __restrict__ out, int nReal, int nTot) {
    int i = blockIdx.x * blockDim.x + threadIdx.x;
    if (i < nReal)      out[i] = f32_to_bf16_rne(in[i]);
    else if (i < nTot)  out[i] = 0;
}

// W [K x Ncols] f32 row-major -> Wt [Ncols x K] bf16 row-major
__global__ void k_transpose_bf16(const float* __restrict__ W,
                                 uint16_t* __restrict__ Wt, int K, int Ncols) {
    int i = blockIdx.x * blockDim.x + threadIdx.x;
    if (i < K * Ncols) {
        int k = i / Ncols, n = i - k * Ncols;
        Wt[(size_t)n * K + k] = f32_to_bf16_rne(W[i]);
    }
}

__global__ void k_zero(float* __restrict__ p, int n) {
    int i = blockIdx.x * blockDim.x + threadIdx.x;
    if (i < n) p[i] = 0.0f;
}

__global__ void k_relu(const float* __restrict__ in, float* __restrict__ out, int n) {
    int i = blockIdx.x * blockDim.x + threadIdx.x;
    if (i < n) out[i] = fmaxf(in[i], 0.0f);
}

// ---------------- dense GEMM: Y[MpxN] = X[MpxK](bf16) @ W (via Wt[NxK]) -----
// Fragment layouts per CDNA5 ISA 7.12.2 (16-bit A 16x32, B 32x16,
// f32 C/D). Mp%64==0, K%32==0, N%128==0 -> exact grid, EXEC all-ones.

__global__ __launch_bounds__(256) void k_gemm_bf16_wmma(
    const uint16_t* __restrict__ X,    // [Mp x K] bf16 row-major
    const uint16_t* __restrict__ Wt,   // [N  x K] bf16 row-major (W transposed)
    float* __restrict__ Y,             // [Mp x N] f32
    int K, int Nn)
{
    __shared__ uint16_t tileA[2][64 * 32];   // ping-pong, 2 x 4 KB

    const int tid   = threadIdx.x;
    const int wave  = tid >> 5;
    const int lane  = tid & 31;
    const int waveM = wave >> 2;                    // 0..1 -> 32-row slab
    const int waveN = wave & 3;                     // 0..3 -> 32-col slab
    const int mTile = blockIdx.y * 64;
    const int nBase = blockIdx.x * 128 + waveN * 32;

    const int half = lane >> 4;      // 0: lanes 0-15, 1: lanes 16-31
    const int l16  = lane & 15;

    f32x8 acc00 = {}, acc01 = {}, acc10 = {}, acc11 = {};

    // A-tile staging: 256 threads x 16 B (b128) = 4 KB tile [64 rows x 32 K]
    const int sRow = tid >> 2;
    const int sCol = (tid & 3) * 8;
    const uint16_t* aSrc = X + (size_t)(mTile + sRow) * K + sCol;
    const int sOff = sRow * 32 + sCol;

    // A fragment LDS offsets (per buffer) for the wave's two 16-row subtiles
    const int aOff0 = (waveM * 32 + 0  + l16) * 32 + half * 8;
    const int aOff1 = (waveM * 32 + 16 + l16) * 32 + half * 8;

    // B fragment base pointers: two 16-byte chunks per lane per subtile (k = 8h..8h+7, 16+8h..16+8h+7)
    const uint16_t* bCol0 = Wt + (size_t)(nBase + 0  + l16) * K + half * 8;
    const uint16_t* bCol1 = Wt + (size_t)(nBase + 16 + l16) * K + half * 8;

    const int T = K >> 5;

    // preload K-tile 0
    *(u32x4*)&tileA[0][sOff] = *(const u32x4*)aSrc;
    __syncthreads();

    for (int t = 0; t < T; ++t) {
        const int kk = t << 5;
        const uint16_t* curA = tileA[t & 1];
        const bool hasNext = (t + 1 < T);

        // stage next A tile into registers (overlaps with the 4 WMMAs below)
        u32x4 stage = {};
        if (hasNext) {
            stage = *(const u32x4*)(aSrc + kk + 32);
            __builtin_prefetch(aSrc + kk + 64, 0, 3);   // global_prefetch_b8
        }

        // A fragments: 4x ds_load_b128
        u32x4 a0lo = *(const u32x4*)&curA[aOff0];
        u32x4 a0hi = *(const u32x4*)&curA[aOff0 + 16];
        u32x4 a1lo = *(const u32x4*)&curA[aOff1];
        u32x4 a1hi = *(const u32x4*)&curA[aOff1 + 16];
        u32x8 av0 = {a0lo[0],a0lo[1],a0lo[2],a0lo[3],a0hi[0],a0hi[1],a0hi[2],a0hi[3]};
        u32x8 av1 = {a1lo[0],a1lo[1],a1lo[2],a1lo[3],a1hi[0],a1hi[1],a1hi[2],a1hi[3]};

        // B fragments: 4x global_load_b128 (L2-resident Wt)
        const u32x4* bp0 = (const u32x4*)(bCol0 + kk);
        const u32x4* bp1 = (const u32x4*)(bCol1 + kk);
        u32x4 b0lo = bp0[0], b0hi = bp0[2];
        u32x4 b1lo = bp1[0], b1hi = bp1[2];
        u32x8 bv0 = {b0lo[0],b0lo[1],b0lo[2],b0lo[3],b0hi[0],b0hi[1],b0hi[2],b0hi[3]};
        u32x8 bv1 = {b1lo[0],b1lo[1],b1lo[2],b1lo[3],b1hi[0],b1hi[1],b1hi[2],b1hi[3]};

        bf16x16 A0 = __builtin_bit_cast(bf16x16, av0);
        bf16x16 A1 = __builtin_bit_cast(bf16x16, av1);
        bf16x16 B0 = __builtin_bit_cast(bf16x16, bv0);
        bf16x16 B1 = __builtin_bit_cast(bf16x16, bv1);

        acc00 = __builtin_amdgcn_wmma_f32_16x16x32_bf16(false, A0, false, B0, (short)0, acc00, false, false);
        acc01 = __builtin_amdgcn_wmma_f32_16x16x32_bf16(false, A0, false, B1, (short)0, acc01, false, false);
        acc10 = __builtin_amdgcn_wmma_f32_16x16x32_bf16(false, A1, false, B0, (short)0, acc10, false, false);
        acc11 = __builtin_amdgcn_wmma_f32_16x16x32_bf16(false, A1, false, B1, (short)0, acc11, false, false);

        // write next tile into the other buffer; safe: readers of that buffer
        // finished before the previous barrier
        if (hasNext)
            *(u32x4*)&tileA[(t + 1) & 1][sOff] = stage;
        __syncthreads();
    }

    // D layout: VGPR r -> (M = r + half*8, N = l16)
    float* y00 = Y + (size_t)(mTile + waveM * 32) * Nn + nBase + l16;
#pragma unroll
    for (int r = 0; r < 8; ++r) {
        const int mr = r + half * 8;
        y00[(size_t)mr * Nn]              = acc00[r];
        y00[(size_t)mr * Nn + 16]         = acc01[r];
        y00[(size_t)(mr + 16) * Nn]       = acc10[r];
        y00[(size_t)(mr + 16) * Nn + 16]  = acc11[r];
    }
}

// ---------------- SpMM: out[row] += val * xw[col]  (COO, H=512) -------------
// 64 threads per edge; float4 (b128) gathers; f32 atomic scatter.

__global__ __launch_bounds__(256) void k_spmm_atomic(
    const int* __restrict__ row, const int* __restrict__ col,
    const float* __restrict__ val, const float* __restrict__ Xw,  // [* x HH]
    float* __restrict__ out, int nnz)
{
    int e = blockIdx.x * 4 + (threadIdx.x >> 6);
    if (e >= nnz) return;
    int g = threadIdx.x & 63;
    int r = row[e], c = col[e];
    float v = val[e];
    const float4* xv = (const float4*)(Xw + (size_t)c * HH);
    float* orow = out + (size_t)r * HH;
#pragma unroll
    for (int jj = 0; jj < 2; ++jj) {
        int j = g + jj * 64;               // 128 float4 per row
        float4 x = xv[j];
        atomicAdd(&orow[j * 4 + 0], v * x.x);
        atomicAdd(&orow[j * 4 + 1], v * x.y);
        atomicAdd(&orow[j * 4 + 2], v * x.z);
        atomicAdd(&orow[j * 4 + 3], v * x.w);
    }
}

// ---------------- final concat: out[n] = [e0 | e0 | e3] ---------------------

__global__ void k_assemble(const float* __restrict__ e00, const float* __restrict__ e01,
                           const float* __restrict__ e30, const float* __restrict__ e31,
                           float* __restrict__ out, int total)
{
    int i = blockIdx.x * blockDim.x + threadIdx.x;
    if (i >= total) return;
    int n = i / (3 * HH), c = i - n * (3 * HH);
    const float *e0, *e3;
    int rr;
    if (n < NN0) { e0 = e00; e3 = e30; rr = n; }
    else         { e0 = e01; e3 = e31; rr = n - NN0; }
    float vo;
    if (c < HH)            vo = e0[(size_t)rr * HH + c];
    else if (c < 2 * HH)   vo = e0[(size_t)rr * HH + (c - HH)];
    else                   vo = e3[(size_t)rr * HH + (c - 2 * HH)];
    out[i] = vo;
}

// ---------------------------------------------------------------------------

extern "C" void kernel_launch(void* const* d_in, const int* in_sizes, int n_in,
                              void* d_out, int out_size, void* d_ws, size_t ws_size,
                              hipStream_t stream)
{
    (void)n_in; (void)out_size; (void)ws_size;

    const float* feat0 = (const float*)d_in[0];
    const float* feat1 = (const float*)d_in[1];

    struct Rel { const int* row; const int* col; const float* val; int nnz; int srcN; int dstN; };
    const int srcN[5] = {NN0, NN1, NN0, NN1, NN1};
    const int dstN[5] = {NN0, NN0, NN1, NN1, NN1};
    Rel rels[5];
    for (int r = 0; r < 5; ++r) {
        rels[r].row  = (const int*)d_in[2 + 3 * r + 0];
        rels[r].col  = (const int*)d_in[2 + 3 * r + 1];
        rels[r].val  = (const float*)d_in[2 + 3 * r + 2];
        rels[r].nnz  = in_sizes[2 + 3 * r + 0];
        rels[r].srcN = srcN[r];
        rels[r].dstN = dstN[r];
    }
    const float* W1[5]; const float* W2[5]; const float* W3[5];
    for (int r = 0; r < 5; ++r) {
        W1[r] = (const float*)d_in[17 + r];
        W2[r] = (const float*)d_in[22 + r];
        W3[r] = (const float*)d_in[27 + r];
    }

    // bump allocator over workspace (~396 MB), fully rewritten every call
    char* ws = (char*)d_ws;
    auto alloc = [&](size_t bytes) -> void* {
        void* p = (void*)ws;
        ws += (bytes + 255) & ~(size_t)255;
        return p;
    };
    uint16_t* Xbf0 = (uint16_t*)alloc((size_t)M0P * FF * 2);
    uint16_t* Xbf1 = (uint16_t*)alloc((size_t)M1P * FF * 2);
    uint16_t* Ebf0 = (uint16_t*)alloc((size_t)M0P * HH * 2);
    uint16_t* Ebf1 = (uint16_t*)alloc((size_t)M1P * HH * 2);
    uint16_t* Wt   = (uint16_t*)alloc((size_t)FF * HH * 2);
    float* xw   = (float*)alloc((size_t)M0P * HH * 4);   // padded rows, reused per relation
    float* acc0 = (float*)alloc((size_t)NN0 * HH * 4);
    float* acc1 = (float*)alloc((size_t)NN1 * HH * 4);
    float* e0_0 = (float*)alloc((size_t)NN0 * HH * 4);
    float* e0_1 = (float*)alloc((size_t)NN1 * HH * 4);
    float* e2_0 = (float*)alloc((size_t)NN0 * HH * 4);
    float* e2_1 = (float*)alloc((size_t)NN1 * HH * 4);

    auto cvt = [&](const float* in, uint16_t* outp, size_t nReal, size_t nTot) {
        k_f32_to_bf16<<<(unsigned)((nTot + 255) / 256), 256, 0, stream>>>(
            in, outp, (int)nReal, (int)nTot);
    };
    auto zero = [&](float* p, size_t n) {
        k_zero<<<(unsigned)((n + 255) / 256), 256, 0, stream>>>(p, (int)n);
    };
    auto gemm = [&](const uint16_t* Xb, const float* Wf, int K, int Mp, float* Yout) {
        size_t we = (size_t)K * HH;
        k_transpose_bf16<<<(unsigned)((we + 255) / 256), 256, 0, stream>>>(Wf, Wt, K, HH);
        dim3 grid(HH / 128, Mp / 64);
        k_gemm_bf16_wmma<<<grid, 256, 0, stream>>>(Xb, Wt, Yout, K, HH);
    };
    auto spmm = [&](const Rel& rl, const float* Xw, float* outAcc) {
        k_spmm_atomic<<<(unsigned)((rl.nnz + 3) / 4), 256, 0, stream>>>(
            rl.row, rl.col, rl.val, Xw, outAcc, rl.nnz);
    };
    auto relu = [&](const float* in, float* outp, size_t n) {
        k_relu<<<(unsigned)((n + 255) / 256), 256, 0, stream>>>(in, outp, (int)n);
    };

    auto layer = [&](const uint16_t* src0, const uint16_t* src1, int K,
                     const float* const* Wl, float* a0, float* a1) {
        zero(a0, (size_t)NN0 * HH);
        zero(a1, (size_t)NN1 * HH);
        for (int r = 0; r < 5; ++r) {
            const bool fromType0 = (rels[r].srcN == NN0);
            gemm(fromType0 ? src0 : src1, Wl[r], K, fromType0 ? M0P : M1P, xw);
            spmm(rels[r], xw, (rels[r].dstN == NN0) ? a0 : a1);
        }
    };

    // one-time bf16 conversion of the raw features (pad rows zero-filled)
    cvt(feat0, Xbf0, (size_t)NN0 * FF, (size_t)M0P * FF);
    cvt(feat1, Xbf1, (size_t)NN1 * FF, (size_t)M1P * FF);

    // Layer 1 (K = 1024) + ReLU
    layer(Xbf0, Xbf1, FF, W1, acc0, acc1);
    relu(acc0, e0_0, (size_t)NN0 * HH);
    relu(acc1, e0_1, (size_t)NN1 * HH);

    // Layer 2 (K = 512) + ReLU
    cvt(e0_0, Ebf0, (size_t)NN0 * HH, (size_t)M0P * HH);
    cvt(e0_1, Ebf1, (size_t)NN1 * HH, (size_t)M1P * HH);
    layer(Ebf0, Ebf1, HH, W2, acc0, acc1);
    relu(acc0, e2_0, (size_t)NN0 * HH);
    relu(acc1, e2_1, (size_t)NN1 * HH);

    // Layer 3 (K = 512), no ReLU; results left in acc0/acc1
    cvt(e2_0, Ebf0, (size_t)NN0 * HH, (size_t)M0P * HH);
    cvt(e2_1, Ebf1, (size_t)NN1 * HH, (size_t)M1P * HH);
    layer(Ebf0, Ebf1, HH, W3, acc0, acc1);

    // out = [e0 | e0 | e3] stacked over node types
    size_t total = (size_t)(NN0 + NN1) * 3 * HH;
    k_assemble<<<(unsigned)((total + 255) / 256), 256, 0, stream>>>(
        e0_0, e0_1, acc0, acc1, (float*)d_out, (int)total);
}
